// RAG_DyGKT_69415261438456
// MI455X (gfx1250) — matrix operand order repaired and checked by reading, not compiled, not run on hardware
//
#include <hip/hip_runtime.h>
#include <hip/hip_bf16.h>
#include <math.h>

// ---------------------------------------------------------------------------
// RAG_DyGKT forward for MI455X (gfx1250, wave32, WMMA).
// All GEMMs via v_wmma_f32_16x16x32_f16 (f32 -> f16 inputs, f32 accumulate).
// ---------------------------------------------------------------------------

typedef __attribute__((ext_vector_type(16))) _Float16 v16h;
typedef __attribute__((ext_vector_type(8)))  float    v8f;

#define DEV __device__ __forceinline__

constexpr int cB  = 512;
constexpr int cLH = 200;
constexpr int cN  = 40000;
constexpr int cF  = 256;
constexpr int cEF = 4;
constexpr int cK  = 8;
constexpr int cLP = 50;
constexpr int cD  = 64;
constexpr int ROWS = cB * cLH;                 // 102400 rows per branch
constexpr int TILES_PER_BRANCH = ROWS / 16;    // 6400
constexpr unsigned cBIG = 1600000000u;         // N*N

DEV v8f wmma16(v16h a, v16h b, v8f c) {
  // emits v_wmma_f32_16x16x32_f16
  return __builtin_amdgcn_wmma_f32_16x16x32_f16(false, a, false, b, (short)0, c,
                                                false, false);
}

// intra-wave LDS fence (CDNA5 split counter); also a compiler memory barrier
DEV void lds_fence() { asm volatile("s_wait_dscnt 0" ::: "memory"); }

// Build A-operand (16x32 f16 tile) for this lane from 32 consecutive f32.
// A layout: half j -> K = (j&7) + (j>=8?16:0) + (lane>=16?8:0)
// -> two contiguous 8-float runs at +hi*8 and +16+hi*8.
DEV v16h pack_a_f32(const float* p, int hi) {
  const float* c1 = p + (hi ? 8 : 0);
  const float* c2 = c1 + 16;
  v16h a;
#pragma unroll
  for (int j = 0; j < 8; ++j) {
    a[j]     = (_Float16)c1[j];
    a[j + 8] = (_Float16)c2[j];
  }
  return a;
}

DEV float sigm(float x) { return 1.f / (1.f + __expf(-x)); }

// ---------------------------------------------------------------------------
// Pack a row-major weight W[outDim][inDim] (f32) into WMMA B-operand tiles:
// dst[((nt*KC+kc)*32+lane)*16+j] = W[nt*16+(lane&15)][kc*32 + (j&7) + (j>=8?16:0) + (lane>=16?8:0)]
// (zero-padded out of range). B layout matches A: half j -> K = (j&7) + (j>=8?16:0)
// + (lane>=16?8:0) within each 32-K chunk.
// ---------------------------------------------------------------------------
__global__ void pack_w(const float* __restrict__ W, _Float16* __restrict__ dst,
                       int outDim, int inDim, int ntiles, int kchunks) {
  int idx = blockIdx.x * blockDim.x + threadIdx.x;
  int total = ntiles * kchunks * 512;
  if (idx >= total) return;
  int j = idx & 15;
  int lane = (idx >> 4) & 31;
  int rest = idx >> 9;
  int kc = rest % kchunks;
  int nt = rest / kchunks;
  int n = nt * 16 + (lane & 15);
  int k = kc * 32 + (j & 7) + ((j >= 8) ? 16 : 0) + ((lane >= 16) ? 8 : 0);
  float v = (n < outDim && k < inDim) ? W[n * inDim + k] : 0.f;
  dst[idx] = (_Float16)v;
}

// ---------------------------------------------------------------------------
// Stage A: dygkt encoder for both branches.
// One wave per 16-row tile: gathered node rows -> A operand, K=256 (8 chunks),
// +time-feature WMMA (K=16 zero-padded chunk), scalar epilogue, layernorm.
// Output Y[branch][b*LH+t][64] (f32, post-layernorm).
// ---------------------------------------------------------------------------
__global__ void encode_kernel(
    const float* __restrict__ node_raw, const float* __restrict__ edge_raw,
    const int* __restrict__ nbrL, const int* __restrict__ eidL, const float* __restrict__ tmL,
    const int* __restrict__ nbrD, const int* __restrict__ eidD, const float* __restrict__ tmD,
    const int* __restrict__ dst_ids, const int* __restrict__ src_ids,
    const _Float16* __restrict__ wF16, const _Float16* __restrict__ wT16,
    const float* __restrict__ bfeat, const float* __restrict__ Wedge,
    const float* __restrict__ bedge, const float* __restrict__ btime,
    const float* __restrict__ Wstruct, const float* __restrict__ bstruct,
    const float* __restrict__ ln_g, const float* __restrict__ ln_b,
    float* __restrict__ Y) {
  int tile = blockIdx.x;
  int branch = (tile >= TILES_PER_BRANCH) ? 1 : 0;
  int rowbase = (tile - branch * TILES_PER_BRANCH) * 16;
  int lane = threadIdx.x;
  int hi = lane >> 4, hi8 = hi * 8, N0 = lane & 15;

  __shared__ float s_co[16], s_sim[16], s_er[16], s_y[16 * 64];

  int r = rowbase + N0;
  int b = r / cLH, t = r - b * cLH;
  const int*   nbr  = branch ? nbrD : nbrL;
  const int*   eids = branch ? eidD : eidL;
  const float* tms  = branch ? tmD  : tmL;
  const int*   curp = branch ? src_ids : dst_ids;
  int   nid = nbr[b * cLH + t];
  int   eid = eids[b * cLH + t];
  float tm  = tms[b * cLH + t];
  int   cur = curp[b];
  if (hi == 0) {
    s_co[N0] = (nid == cur) ? 1.f : 0.f;
    int csk = (int)node_raw[(size_t)cur * cF];
    s_sim[N0] = ((int)node_raw[(size_t)nid * cF] == csk) ? 1.f : 0.f;
    s_er[N0] = edge_raw[(size_t)eid * cEF];
  }

  const v16h* WF = (const v16h*)wF16;
  const v16h* WT = (const v16h*)wT16;
  v8f acc[4] = {{}, {}, {}, {}};
  const float* rowp = node_raw + (size_t)nid * cF;
#pragma unroll
  for (int kc = 0; kc < 8; ++kc) {
    v16h a = pack_a_f32(rowp + kc * 32, hi);
#pragma unroll
    for (int nt = 0; nt < 4; ++nt)
      acc[nt] = wmma16(a, WF[(nt * 8 + kc) * 32 + lane], acc[nt]);
  }
  // time features: K = 0..15 real, 16..31 zero-padded (weights packed with zeros)
  v16h at;
#pragma unroll
  for (int j = 0; j < 8; ++j) {
    int kk = j + hi8;
    float fr = exp10f(-(float)kk * (5.f / 16.f));
    at[j]     = (_Float16)cosf(tm * fr);
    at[j + 8] = (_Float16)0.f;
  }
#pragma unroll
  for (int nt = 0; nt < 4; ++nt)
    acc[nt] = wmma16(at, WT[nt * 32 + lane], acc[nt]);

  __syncthreads();
#pragma unroll
  for (int nt = 0; nt < 4; ++nt)
#pragma unroll
    for (int e = 0; e < 8; ++e) {
      int M2 = e + hi8, d = nt * 16 + N0;
      float val = acc[nt][e] + bfeat[d] + btime[d]
                + s_er[M2] * Wedge[d] + bedge[d]
                + (s_co[M2] + s_sim[M2]) * Wstruct[d] + 2.f * bstruct[d];
      s_y[M2 * 64 + d] = val;
    }
  __syncthreads();
  if (hi == 0) {
    float m = 0.f;
#pragma unroll
    for (int d = 0; d < 64; ++d) m += s_y[N0 * 64 + d];
    m *= (1.f / 64.f);
    float v = 0.f;
#pragma unroll
    for (int d = 0; d < 64; ++d) { float x = s_y[N0 * 64 + d] - m; v += x * x; }
    v *= (1.f / 64.f);
    float rs = rsqrtf(v + 1e-5f);
    size_t ob = ((size_t)branch * ROWS + (size_t)(rowbase + N0)) * 64;
#pragma unroll
    for (int d = 0; d < 64; ++d)
      Y[ob + d] = (s_y[N0 * 64 + d] - m) * rs * ln_g[d] + ln_b[d];
  }
}

// ---------------------------------------------------------------------------
// graph_emb: one 256-thread workgroup per batch element. Everything lives in
// dynamic LDS (~240KB of the WGP's 320KB): sorted ids, edge keys, h (400x64),
// agg (400x64). Two chained WMMA GEMMs per 16-row tile.
// ---------------------------------------------------------------------------
DEV int lb512(const int* a, int val) {
  int lo = 0, hh = 512;
  while (lo < hh) { int mid = (lo + hh) >> 1; if (a[mid] < val) lo = mid + 1; else hh = mid; }
  return lo;
}

template <typename T>
DEV void bitonic512(T* a, int tid) {
  for (int k = 2; k <= 512; k <<= 1)
    for (int j = k >> 1; j > 0; j >>= 1) {
      __syncthreads();
      for (int i = tid; i < 512; i += 256) {
        int ixj = i ^ j;
        if (ixj > i) {
          T x = a[i], y = a[ixj];
          bool up = ((i & k) == 0);
          if ((x > y) == up) { a[i] = y; a[ixj] = x; }
        }
      }
    }
  __syncthreads();
}

DEV float uniqf(const int* sid, int i) {
  return (sid[i] < cN && (i == 0 || sid[i - 1] != sid[i])) ? 1.f : 0.f;
}

__global__ void graph_kernel(const int* __restrict__ pool_nodes, const int* __restrict__ retr,
                             const float* __restrict__ node_raw,
                             const _Float16* __restrict__ wF16, const _Float16* __restrict__ gcn16,
                             const float* __restrict__ bfeat, const float* __restrict__ gcn_b,
                             float* __restrict__ glob) {
  extern __shared__ char smem[];
  int*      s_demo = (int*)(smem + 0);        // 400 ints
  int*      s_sid  = (int*)(smem + 1600);     // 512 ints (sorted, sentinel = N)
  unsigned* s_key  = (unsigned*)(smem + 3648);// 512 edge keys
  float*    s_deg  = (float*)(smem + 5696);   // 400 (deg, then dis)
  float*    s_red  = (float*)(smem + 7296);   // 4*64 reduction
  int*      s_cnt  = (int*)(smem + 8320);
  float*    s_t    = (float*)(smem + 8352);   // 8 waves * 16*64 scratch
  float*    s_h    = (float*)(smem + 41120);  // 400*64
  float*    s_agg  = (float*)(smem + 143520); // 400*64

  int tid = threadIdx.x;
  int b = blockIdx.x;

  // 1. load demo ids; sentinel-pad for sorting
  for (int i = tid; i < 512; i += 256) {
    int id = 0;
    if (i < 400) {
      int g = retr[b * cK + i / cLP];
      id = pool_nodes[g * cLP + (i % cLP)];
      s_demo[i] = id;
    }
    s_sid[i] = (i < 400 && id > 0) ? id : cN;
  }
  bitonic512(s_sid, tid);

  // 2. per-wave chained GEMMs: t = node_raw[sid]@Wfeat.T + bfeat ; h = (t@gcnW.T)*uniq
  int wv = tid >> 5, lane = tid & 31;
  int hi = lane >> 4, hi8 = hi * 8, N0 = lane & 15;
  const v16h* WF = (const v16h*)wF16;
  const v16h* GW = (const v16h*)gcn16;
  float* myT = s_t + wv * (16 * 64);
  for (int tile = wv; tile < 25; tile += 8) {
    int rowbase = tile * 16;
    int i = rowbase + N0;
    int id = s_sid[i]; if (id > cN - 1) id = cN - 1;   // clip sentinel
    v8f a0 = {}, a1 = {}, a2 = {}, a3 = {};
    const float* rowp = node_raw + (size_t)id * cF;
#pragma unroll
    for (int kc = 0; kc < 8; ++kc) {
      v16h av = pack_a_f32(rowp + kc * 32, hi);
      a0 = wmma16(av, WF[(0 * 8 + kc) * 32 + lane], a0);
      a1 = wmma16(av, WF[(1 * 8 + kc) * 32 + lane], a1);
      a2 = wmma16(av, WF[(2 * 8 + kc) * 32 + lane], a2);
      a3 = wmma16(av, WF[(3 * 8 + kc) * 32 + lane], a3);
    }
#pragma unroll
    for (int e = 0; e < 8; ++e) {
      int M2 = e + hi8;
      myT[M2 * 64 +  0 + N0] = a0[e] + bfeat[ 0 + N0];
      myT[M2 * 64 + 16 + N0] = a1[e] + bfeat[16 + N0];
      myT[M2 * 64 + 32 + N0] = a2[e] + bfeat[32 + N0];
      myT[M2 * 64 + 48 + N0] = a3[e] + bfeat[48 + N0];
    }
    lds_fence();
    v8f h0 = {}, h1 = {}, h2 = {}, h3 = {};
#pragma unroll
    for (int kc = 0; kc < 2; ++kc) {
      v16h av = pack_a_f32(myT + N0 * 64 + kc * 32, hi);
      h0 = wmma16(av, GW[(0 * 2 + kc) * 32 + lane], h0);
      h1 = wmma16(av, GW[(1 * 2 + kc) * 32 + lane], h1);
      h2 = wmma16(av, GW[(2 * 2 + kc) * 32 + lane], h2);
      h3 = wmma16(av, GW[(3 * 2 + kc) * 32 + lane], h3);
    }
#pragma unroll
    for (int e = 0; e < 8; ++e) {
      int i2 = rowbase + e + hi8;
      float uf = uniqf(s_sid, i2);
      s_h[i2 * 64 +  0 + N0] = h0[e] * uf;
      s_h[i2 * 64 + 16 + N0] = h1[e] * uf;
      s_h[i2 * 64 + 32 + N0] = h2[e] * uf;
      s_h[i2 * 64 + 48 + N0] = h3[e] * uf;
    }
    lds_fence();
  }
  __syncthreads();

  // 3. edge keys from consecutive pairs (original demo order), sort, dedup
  for (int e = tid; e < 512; e += 256) {
    unsigned key = cBIG;
    if (e < 392) {
      int p = e / 49, q = e - p * 49;
      int u = s_demo[p * 50 + q], v = s_demo[p * 50 + q + 1];
      if (u > 0 && v > 0) {
        unsigned aa = (unsigned)(u < v ? u : v);
        unsigned bb = (unsigned)(u < v ? v : u);
        key = aa * (unsigned)cN + bb;
      }
    }
    s_key[e] = key;
  }
  bitonic512(s_key, tid);

  for (int i = tid; i < 400; i += 256) s_deg[i] = 0.f;
  for (int i = tid; i < 400 * 64; i += 256) s_agg[i] = 0.f;
  if (tid == 0) *s_cnt = 0;
  __syncthreads();

  // 4. degrees (+ self loop), dis = rsqrt(deg), unique count
  for (int e = tid; e < 512; e += 256) {
    unsigned key = s_key[e];
    if (key < cBIG && (e == 0 || s_key[e - 1] != key)) {
      int su = lb512(s_sid, (int)(key / (unsigned)cN)); if (su > 399) su = 399;
      int sv = lb512(s_sid, (int)(key % (unsigned)cN)); if (sv > 399) sv = 399;
      atomicAdd(&s_deg[su], 1.f);
      atomicAdd(&s_deg[sv], 1.f);
    }
  }
  __syncthreads();
  for (int i = tid; i < 400; i += 256) {
    float uf = uniqf(s_sid, i);
    float dg = s_deg[i] + uf;
    s_deg[i] = (dg > 0.f) ? rsqrtf(dg) : 0.f;  // now holds dis
    if (uf > 0.f) atomicAdd(s_cnt, 1);
  }
  __syncthreads();

  // 5. normalized neighbor aggregation (64 lanes per edge slot-group)
  int dcol = tid & 63, eg = tid >> 6;
  for (int e = eg; e < 512; e += 4) {
    unsigned key = s_key[e];
    if (key < cBIG && (e == 0 || s_key[e - 1] != key)) {
      int su = lb512(s_sid, (int)(key / (unsigned)cN)); if (su > 399) su = 399;
      int sv = lb512(s_sid, (int)(key % (unsigned)cN)); if (sv > 399) sv = 399;
      float nrm = s_deg[su] * s_deg[sv];
      atomicAdd(&s_agg[sv * 64 + dcol], nrm * s_h[su * 64 + dcol]);
      atomicAdd(&s_agg[su * 64 + dcol], nrm * s_h[sv * 64 + dcol]);
    }
  }
  __syncthreads();

  // 6. + self term + bias, relu, mean over unique nodes
  float part = 0.f;
  for (int i = eg; i < 400; i += 4) {
    float dis = s_deg[i];
    float uf = uniqf(s_sid, i);
    float v = s_agg[i * 64 + dcol] + s_h[i * 64 + dcol] * dis * dis + gcn_b[dcol] * uf;
    part += fmaxf(v, 0.f);
  }
  s_red[eg * 64 + dcol] = part;
  __syncthreads();
  if (tid < 64) {
    float tot = s_red[tid] + s_red[64 + tid] + s_red[128 + tid] + s_red[192 + tid];
    int c = *s_cnt; if (c < 1) c = 1;
    glob[(size_t)b * 64 + tid] = tot / (float)c;
  }
}

// ---------------------------------------------------------------------------
// GRU scan: recurrence is independent per batch row, so one wave owns 16 rows
// for all 200 steps (no global sync). Per step: h -> LDS -> A operand,
// x from Y, 48 WMMAs (r/z merged gi+gh; n kept split), gate math in VALU.
// grid = 64 (2 GRUs x 32 row tiles).
// ---------------------------------------------------------------------------
__global__ void gru_kernel(const float* __restrict__ Y,
                           const _Float16* __restrict__ wihL, const _Float16* __restrict__ whhL,
                           const _Float16* __restrict__ wihQ, const _Float16* __restrict__ whhQ,
                           const float* __restrict__ bihL, const float* __restrict__ bhhL,
                           const float* __restrict__ bihQ, const float* __restrict__ bhhQ,
                           float* __restrict__ Hout) {
  int wg = blockIdx.x;
  int branch = wg >> 5;
  int rowbase = (wg & 31) * 16;
  int lane = threadIdx.x;
  int hi = lane >> 4, hi8 = hi * 8, N0 = lane & 15;
  const v16h* WIH = (const v16h*)(branch ? wihQ : wihL);
  const v16h* WHH = (const v16h*)(branch ? whhQ : whhL);
  const float* bih = branch ? bihQ : bihL;
  const float* bhh = branch ? bhhQ : bhhL;

  float br_[4], bz_[4], bn_[4], hr_[4], hz_[4], hn_[4];
#pragma unroll
  for (int g = 0; g < 4; ++g) {
    int d = g * 16 + N0;
    br_[g] = bih[d];      bz_[g] = bih[64 + d]; bn_[g] = bih[128 + d];
    hr_[g] = bhh[d];      hz_[g] = bhh[64 + d]; hn_[g] = bhh[128 + d];
  }

  v8f hC[4] = {{}, {}, {}, {}};
  __shared__ float hbuf[16 * 64];
  int bRow = rowbase + N0;
  const float* ybase = Y + ((size_t)branch * ROWS + (size_t)bRow * cLH) * 64;

  for (int t = 0; t < cLH; ++t) {
#pragma unroll
    for (int g = 0; g < 4; ++g)
#pragma unroll
      for (int e = 0; e < 8; ++e)
        hbuf[(e + hi8) * 64 + g * 16 + N0] = hC[g][e];
    __syncthreads();

    v16h ah0 = pack_a_f32(&hbuf[N0 * 64 + 0], hi);
    v16h ah1 = pack_a_f32(&hbuf[N0 * 64 + 32], hi);
    const float* yrow = ybase + (size_t)t * 64;
    v16h ax0 = pack_a_f32(yrow, hi);
    v16h ax1 = pack_a_f32(yrow + 32, hi);
    if (t + 1 < cLH) __builtin_prefetch(yrow + 64, 0, 1);  // global_prefetch_b8

#pragma unroll
    for (int g = 0; g < 4; ++g) {
      int nR = g, nZ = g + 4, nN = g + 8;
      v8f aR = {}, aZ = {}, aXN = {}, aHN = {};
      aR  = wmma16(ax0, WIH[(nR * 2 + 0) * 32 + lane], aR);
      aR  = wmma16(ax1, WIH[(nR * 2 + 1) * 32 + lane], aR);
      aR  = wmma16(ah0, WHH[(nR * 2 + 0) * 32 + lane], aR);
      aR  = wmma16(ah1, WHH[(nR * 2 + 1) * 32 + lane], aR);
      aZ  = wmma16(ax0, WIH[(nZ * 2 + 0) * 32 + lane], aZ);
      aZ  = wmma16(ax1, WIH[(nZ * 2 + 1) * 32 + lane], aZ);
      aZ  = wmma16(ah0, WHH[(nZ * 2 + 0) * 32 + lane], aZ);
      aZ  = wmma16(ah1, WHH[(nZ * 2 + 1) * 32 + lane], aZ);
      aXN = wmma16(ax0, WIH[(nN * 2 + 0) * 32 + lane], aXN);
      aXN = wmma16(ax1, WIH[(nN * 2 + 1) * 32 + lane], aXN);
      aHN = wmma16(ah0, WHH[(nN * 2 + 0) * 32 + lane], aHN);
      aHN = wmma16(ah1, WHH[(nN * 2 + 1) * 32 + lane], aHN);
#pragma unroll
      for (int e = 0; e < 8; ++e) {
        float rr = sigm(aR[e] + br_[g] + hr_[g]);
        float zz = sigm(aZ[e] + bz_[g] + hz_[g]);
        float nn = tanhf(aXN[e] + bn_[g] + rr * (aHN[e] + hn_[g]));
        hC[g][e] = (1.f - zz) * nn + zz * hC[g][e];
      }
    }
    __syncthreads();
  }
#pragma unroll
  for (int g = 0; g < 4; ++g)
#pragma unroll
    for (int e = 0; e < 8; ++e)
      Hout[((size_t)branch * cB + (size_t)(rowbase + e + hi8)) * 64 + g * 16 + N0] = hC[g][e];
}

// ---------------------------------------------------------------------------
// Head: fused = relu([local_emb, glob] @ fus_W.T + fus_b); out1 = fused@outW.T+b;
// out2 = q_emb@outW.T+b. One wave per 16-row tile.
// ---------------------------------------------------------------------------
__global__ void head_kernel(const float* __restrict__ HL, const float* __restrict__ HQ,
                            const float* __restrict__ GLB,
                            const _Float16* __restrict__ fus16, const _Float16* __restrict__ outW16,
                            const float* __restrict__ fus_b, const float* __restrict__ out_b,
                            float* __restrict__ out) {
  int rowbase = blockIdx.x * 16;
  int lane = threadIdx.x;
  int hi = lane >> 4, hi8 = hi * 8, N0 = lane & 15;
  const v16h* FW = (const v16h*)fus16;
  const v16h* OW = (const v16h*)outW16;
  __shared__ float fbuf[16 * 64];
  int row = rowbase + N0;

  v8f f0 = {}, f1 = {}, f2 = {}, f3 = {};
#pragma unroll
  for (int kc = 0; kc < 4; ++kc) {
    const float* src = (kc < 2) ? (HL + (size_t)row * 64 + kc * 32)
                                : (GLB + (size_t)row * 64 + (kc - 2) * 32);
    v16h av = pack_a_f32(src, hi);
    f0 = wmma16(av, FW[(0 * 4 + kc) * 32 + lane], f0);
    f1 = wmma16(av, FW[(1 * 4 + kc) * 32 + lane], f1);
    f2 = wmma16(av, FW[(2 * 4 + kc) * 32 + lane], f2);
    f3 = wmma16(av, FW[(3 * 4 + kc) * 32 + lane], f3);
  }
#pragma unroll
  for (int e = 0; e < 8; ++e) {
    int M2 = e + hi8;
    fbuf[M2 * 64 +  0 + N0] = fmaxf(f0[e] + fus_b[ 0 + N0], 0.f);
    fbuf[M2 * 64 + 16 + N0] = fmaxf(f1[e] + fus_b[16 + N0], 0.f);
    fbuf[M2 * 64 + 32 + N0] = fmaxf(f2[e] + fus_b[32 + N0], 0.f);
    fbuf[M2 * 64 + 48 + N0] = fmaxf(f3[e] + fus_b[48 + N0], 0.f);
  }
  __syncthreads();

  v8f o0 = {}, o1 = {}, o2 = {}, o3 = {};
  v8f q0 = {}, q1 = {}, q2 = {}, q3 = {};
#pragma unroll
  for (int kc = 0; kc < 2; ++kc) {
    v16h af = pack_a_f32(fbuf + N0 * 64 + kc * 32, hi);
    v16h aq = pack_a_f32(HQ + (size_t)row * 64 + kc * 32, hi);
    o0 = wmma16(af, OW[(0 * 2 + kc) * 32 + lane], o0);
    o1 = wmma16(af, OW[(1 * 2 + kc) * 32 + lane], o1);
    o2 = wmma16(af, OW[(2 * 2 + kc) * 32 + lane], o2);
    o3 = wmma16(af, OW[(3 * 2 + kc) * 32 + lane], o3);
    q0 = wmma16(aq, OW[(0 * 2 + kc) * 32 + lane], q0);
    q1 = wmma16(aq, OW[(1 * 2 + kc) * 32 + lane], q1);
    q2 = wmma16(aq, OW[(2 * 2 + kc) * 32 + lane], q2);
    q3 = wmma16(aq, OW[(3 * 2 + kc) * 32 + lane], q3);
  }
#pragma unroll
  for (int e = 0; e < 8; ++e) {
    int M2 = e + hi8;
    size_t rb = (size_t)(rowbase + M2) * 64;
    out[rb +  0 + N0] = o0[e] + out_b[ 0 + N0];
    out[rb + 16 + N0] = o1[e] + out_b[16 + N0];
    out[rb + 32 + N0] = o2[e] + out_b[32 + N0];
    out[rb + 48 + N0] = o3[e] + out_b[48 + N0];
    out[32768 + rb +  0 + N0] = q0[e] + out_b[ 0 + N0];
    out[32768 + rb + 16 + N0] = q1[e] + out_b[16 + N0];
    out[32768 + rb + 32 + N0] = q2[e] + out_b[32 + N0];
    out[32768 + rb + 48 + N0] = q3[e] + out_b[48 + N0];
  }
}

// ---------------------------------------------------------------------------
extern "C" void kernel_launch(void* const* d_in, const int* in_sizes, int n_in,
                              void* d_out, int out_size, void* d_ws, size_t ws_size,
                              hipStream_t stream) {
  (void)in_sizes; (void)n_in; (void)out_size; (void)ws_size;
  const int*   src_ids  = (const int*)d_in[0];
  const int*   dst_ids  = (const int*)d_in[1];
  const int*   nbrL     = (const int*)d_in[2];
  const int*   eidL     = (const int*)d_in[3];
  const float* tmL      = (const float*)d_in[4];
  const int*   nbrD     = (const int*)d_in[5];
  const int*   eidD     = (const int*)d_in[6];
  const float* tmD      = (const float*)d_in[7];
  const int*   retr     = (const int*)d_in[8];
  const float* node_raw = (const float*)d_in[9];
  const float* edge_raw = (const float*)d_in[10];
  const int*   pool     = (const int*)d_in[11];
  const float* Wfeat    = (const float*)d_in[12];
  const float* bfeat    = (const float*)d_in[13];
  const float* Wedge    = (const float*)d_in[14];
  const float* bedge    = (const float*)d_in[15];
  const float* Wtime    = (const float*)d_in[16];
  const float* btime    = (const float*)d_in[17];
  const float* Wstruct  = (const float*)d_in[18];
  const float* bstruct  = (const float*)d_in[19];
  const float* ln_g     = (const float*)d_in[20];
  const float* ln_b     = (const float*)d_in[21];
  const float* WihL     = (const float*)d_in[22];
  const float* WhhL     = (const float*)d_in[23];
  const float* bihL     = (const float*)d_in[24];
  const float* bhhL     = (const float*)d_in[25];
  const float* WihQ     = (const float*)d_in[26];
  const float* WhhQ     = (const float*)d_in[27];
  const float* bihQ     = (const float*)d_in[28];
  const float* bhhQ     = (const float*)d_in[29];
  const float* gcnW     = (const float*)d_in[30];
  const float* gcnb     = (const float*)d_in[31];
  const float* fusW     = (const float*)d_in[32];
  const float* fusb     = (const float*)d_in[33];
  const float* outW     = (const float*)d_in[34];
  const float* outb     = (const float*)d_in[35];

  // workspace layout: packed f16 weights, then f32 activations
  _Float16* wF16   = (_Float16*)d_ws;
  _Float16* wT16   = wF16   + 4 * 8 * 512;
  _Float16* wihL16 = wT16   + 4 * 1 * 512;
  _Float16* whhL16 = wihL16 + 12 * 2 * 512;
  _Float16* wihQ16 = whhL16 + 12 * 2 * 512;
  _Float16* whhQ16 = wihQ16 + 12 * 2 * 512;
  _Float16* gcn16  = whhQ16 + 12 * 2 * 512;
  _Float16* fus16  = gcn16  + 4 * 2 * 512;
  _Float16* out16  = fus16  + 4 * 4 * 512;
  float* Y   = (float*)(out16 + 4 * 2 * 512);
  float* GLB = Y + (size_t)2 * ROWS * 64;       // 512 x 64 graph embeddings
  float* HL  = GLB + (size_t)cB * 64;           // local_emb (branch 0)
  float* HQ  = HL + (size_t)cB * 64;            // q_emb     (branch 1)

  auto pk = [&](const float* W, _Float16* dst, int o, int i, int nt, int kc) {
    int total = nt * kc * 512;
    pack_w<<<(total + 255) / 256, 256, 0, stream>>>(W, dst, o, i, nt, kc);
  };
  pk(Wfeat, wF16,   64, 256,  4, 8);
  pk(Wtime, wT16,   64,  16,  4, 1);
  pk(WihL,  wihL16, 192, 64, 12, 2);
  pk(WhhL,  whhL16, 192, 64, 12, 2);
  pk(WihQ,  wihQ16, 192, 64, 12, 2);
  pk(WhhQ,  whhQ16, 192, 64, 12, 2);
  pk(gcnW,  gcn16,  64,  64,  4, 2);
  pk(fusW,  fus16,  64, 128,  4, 4);
  pk(outW,  out16,  64,  64,  4, 2);

  encode_kernel<<<2 * TILES_PER_BRANCH, 32, 0, stream>>>(
      node_raw, edge_raw, nbrL, eidL, tmL, nbrD, eidD, tmD, dst_ids, src_ids,
      wF16, wT16, bfeat, Wedge, bedge, btime, Wstruct, bstruct, ln_g, ln_b, Y);

  graph_kernel<<<cB, 256, 245920, stream>>>(pool, retr, node_raw, wF16, gcn16,
                                            bfeat, gcnb, GLB);

  gru_kernel<<<64, 32, 0, stream>>>(Y, wihL16, whhL16, wihQ16, whhQ16,
                                    bihL, bhhL, bihQ, bhhQ, HL);

  head_kernel<<<32, 32, 0, stream>>>(HL, HQ, GLB, fus16, out16, fusb, outb,
                                     (float*)d_out);
}
